// Decoder_76768245449551
// MI455X (gfx1250) — matrix operand order repaired and checked by reading, not compiled, not run on hardware
//
#include <hip/hip_runtime.h>
#include <math.h>

// ---------------------------------------------------------------------------
// MI455X (gfx1250) transformer-decoder forward.
// All dense math runs through v_wmma_f32_16x16x32_bf16 (bf16 in, f32 acc).
// Roofline: ~150 GFLOP of GEMM vs ~0.6 GB of mandatory output traffic
// (~25us @ 23.3 TB/s); bf16 WMMA keeps compute comparable to the memory
// floor, while f32 16x16x4 WMMA would be ~8x slower. Every GEMM operand
// fits in the 192MB L2, so fragments are loaded straight from global
// (L2-resident) with per-lane contiguous 16B loads matching the ISA's
// A(16x32)/B(32x16) bf16 VGPR layouts -- no LDS staging needed.
// ---------------------------------------------------------------------------

typedef __bf16 bf16_t;
typedef __attribute__((ext_vector_type(16))) __bf16 v16bf;
typedef __attribute__((ext_vector_type(8)))  __bf16 v8bf;
typedef __attribute__((ext_vector_type(8)))  float   v8f;

#define BB   16
#define LL   1024
#define DD   512
#define NFSZ 256
#define RR   (BB*LL)

__device__ __forceinline__ bf16_t f2bf(float x) {
  union { float f; unsigned u; } v; v.f = x;
  unsigned r = v.u + 0x7fffu + ((v.u >> 16) & 1u);   // round-to-nearest-even
  unsigned short h = (unsigned short)(r >> 16);
  bf16_t o; __builtin_memcpy(&o, &h, 2); return o;
}

// ---------------------------------------------------------------------------
// Batched/strided GEMM: C[M,N] = alpha * A[M,K](bf16) x Bt[N,K](bf16)^T
// + epilogue. grid.z = Z batches, z -> (b = z/H, h = z%H) with separate
// batch/head strides per operand. Wave tile 32x64 (2x4 wmma tiles),
// block = 8 waves -> 128x128. K must be a multiple of 32.
// ---------------------------------------------------------------------------
__global__ __launch_bounds__(256)
void gemm_bf16_wmma(const bf16_t* __restrict__ A,  int ldA, long long sAb, long long sAh,
                    const bf16_t* __restrict__ Bt, int ldB, long long sBb, long long sBh,
                    float* __restrict__ C, int ldC, long long sCb, long long sCh,
                    bf16_t* __restrict__ Cb,
                    int M, int N, int K, int H, float alpha,
                    const float* __restrict__ biasN, const float* __restrict__ biasM,
                    const float* __restrict__ mulT,  const float* __restrict__ addT,
                    const float* __restrict__ pos, int posL, int relu)
{
  const int z    = blockIdx.z;
  const int bidx = z / H;
  const int hidx = z - bidx * H;
  A  += (long long)bidx * sAb + (long long)hidx * sAh;
  Bt += (long long)bidx * sBb + (long long)hidx * sBh;
  const long long cOff = (long long)bidx * sCb + (long long)hidx * sCh;

  const int lane = threadIdx.x & 31;
  const int wave = threadIdx.x >> 5;
  const int m0 = blockIdx.y * 128 + (wave >> 1) * 32;
  const int n0 = blockIdx.x * 128 + (wave & 1) * 64;
  const int l16 = lane & 15;
  const int hb  = lane >> 4;          // lane half: selects K sub-range
  const int kA  = hb * 8;             // A frag: K in {kA..kA+7, kA+16..kA+23}
  const int kB  = hb * 8;             // B frag: K in {kB..kB+7, kB+16..kB+23}

  int ar[2], bc[4];
  ar[0] = m0 + l16;       if (ar[0] > M - 1) ar[0] = M - 1;
  ar[1] = m0 + 16 + l16;  if (ar[1] > M - 1) ar[1] = M - 1;
#pragma unroll
  for (int j = 0; j < 4; ++j) {
    bc[j] = n0 + j * 16 + l16;  if (bc[j] > N - 1) bc[j] = N - 1;
  }

  v8f acc[2][4] = {};
  for (int k0 = 0; k0 < K; k0 += 32) {
    v16bf af[2], bf[4];
#pragma unroll
    for (int i = 0; i < 2; ++i) {
      const bf16_t* p = A + (long long)ar[i] * ldA + k0 + kA;
      v8bf lo = *(const v8bf*)(p);
      v8bf hi = *(const v8bf*)(p + 16);
      af[i] = __builtin_shufflevector(lo, hi, 0,1,2,3,4,5,6,7,8,9,10,11,12,13,14,15);
    }
#pragma unroll
    for (int j = 0; j < 4; ++j) {
      const bf16_t* p = Bt + (long long)bc[j] * ldB + k0 + kB;
      v8bf lo = *(const v8bf*)(p);
      v8bf hi = *(const v8bf*)(p + 16);
      bf[j] = __builtin_shufflevector(lo, hi, 0,1,2,3,4,5,6,7,8,9,10,11,12,13,14,15);
    }
#pragma unroll
    for (int i = 0; i < 2; ++i)
#pragma unroll
      for (int j = 0; j < 4; ++j)
        acc[i][j] = __builtin_amdgcn_wmma_f32_16x16x32_bf16(
            false, af[i], false, bf[j], (short)0, acc[i][j], false, false);
  }

  // Epilogue order (matches model math): alpha, +biasN, +biasM, *mulT, +addT, +pos, relu
#pragma unroll
  for (int i = 0; i < 2; ++i) {
    const int rb = m0 + i * 16 + hb * 8;
#pragma unroll
    for (int j = 0; j < 4; ++j) {
      const int col = n0 + j * 16 + l16;
      if (col < N) {
#pragma unroll
        for (int r = 0; r < 8; ++r) {
          const int row = rb + r;
          if (row < M) {
            const long long ci = cOff + (long long)row * ldC + col;
            float x = acc[i][j][r] * alpha;
            if (biasN) x += biasN[col];
            if (biasM) x += biasM[row];
            if (mulT)  x *= mulT[ci];
            if (addT)  x += addT[ci];
            if (pos)   x += pos[(long long)(row % posL) * DD + col];
            if (relu)  x = fmaxf(x, 0.0f);
            if (C)  C[ci]  = x;
            if (Cb) Cb[ci] = f2bf(x);
          }
        }
      }
    }
  }
}

// ---------------------------------------------------------------------------
// Row softmax over 1024 columns; optional causal mask (k > q -> -1e9 first,
// exactly like the reference). Writes f32 in-place + bf16 mirror.
// ---------------------------------------------------------------------------
__global__ __launch_bounds__(256)
void softmax1024(float* __restrict__ x, bf16_t* __restrict__ xb, int causal)
{
  const long long row = blockIdx.x;
  float* p = x + row * 1024;
  const int q = (int)(row % LL);
  const int t = threadIdx.x;
  float v[4];
  float mx = -3.0e38f;
#pragma unroll
  for (int i = 0; i < 4; ++i) {
    const int c = t + i * 256;
    float s = p[c];
    if (causal && c > q) s = -1.0e9f;
    v[i] = s;
    mx = fmaxf(mx, s);
  }
  __shared__ float red[256];
  red[t] = mx; __syncthreads();
  for (int s = 128; s > 0; s >>= 1) { if (t < s) red[t] = fmaxf(red[t], red[t + s]); __syncthreads(); }
  mx = red[0]; __syncthreads();
  float sum = 0.f;
#pragma unroll
  for (int i = 0; i < 4; ++i) { v[i] = expf(v[i] - mx); sum += v[i]; }
  red[t] = sum; __syncthreads();
  for (int s = 128; s > 0; s >>= 1) { if (t < s) red[t] += red[t + s]; __syncthreads(); }
  const float inv = 1.0f / red[0];
#pragma unroll
  for (int i = 0; i < 4; ++i) {
    const int c = t + i * 256;
    const float o = v[i] * inv;
    p[c] = o;
    xb[row * 1024 + c] = f2bf(o);
  }
}

// ---------------------------------------------------------------------------
// LayerNorm over last dim (512). One block per row; writes f32 + bf16 mirror.
// ---------------------------------------------------------------------------
__global__ __launch_bounds__(256)
void layernorm512(const float* __restrict__ x, const float* __restrict__ g,
                  const float* __restrict__ b, float* __restrict__ y,
                  bf16_t* __restrict__ yb)
{
  const long long row = blockIdx.x;
  const float* p = x + row * DD;
  const int t = threadIdx.x;
  const float a0 = p[t], a1 = p[t + 256];
  __shared__ float red[256];
  red[t] = a0 + a1; __syncthreads();
  for (int s = 128; s > 0; s >>= 1) { if (t < s) red[t] += red[t + s]; __syncthreads(); }
  const float mean = red[0] * (1.0f / 512.0f);
  __syncthreads();
  const float d0 = a0 - mean, d1 = a1 - mean;
  red[t] = d0 * d0 + d1 * d1; __syncthreads();
  for (int s = 128; s > 0; s >>= 1) { if (t < s) red[t] += red[t + s]; __syncthreads(); }
  const float inv = rsqrtf(red[0] * (1.0f / 512.0f) + 1e-5f);
  const float o0 = d0 * inv * g[t] + b[t];
  const float o1 = d1 * inv * g[t + 256] + b[t + 256];
  y[row * DD + t] = o0;       y[row * DD + t + 256] = o1;
  yb[row * DD + t] = f2bf(o0); yb[row * DD + t + 256] = f2bf(o1);
}

// f32 [rows,cols] -> bf16 [cols,rows], batched over grid.z (stride rows*cols).
__global__ __launch_bounds__(256)
void transpose_bf16_kernel(const float* __restrict__ in, long long sIn,
                           bf16_t* __restrict__ out, long long sOut,
                           int rows, int cols)
{
  __shared__ float tile[32][33];
  in  += (long long)blockIdx.z * sIn;
  out += (long long)blockIdx.z * sOut;
  const int c0 = blockIdx.x * 32, r0 = blockIdx.y * 32;
  const int tx = threadIdx.x & 31, ty = threadIdx.x >> 5;   // 32x8
#pragma unroll
  for (int i = 0; i < 32; i += 8) {
    const int r = r0 + ty + i, c = c0 + tx;
    tile[ty + i][tx] = (r < rows && c < cols) ? in[(long long)r * cols + c] : 0.f;
  }
  __syncthreads();
#pragma unroll
  for (int i = 0; i < 32; i += 8) {
    const int c = c0 + ty + i, r = r0 + tx;
    if (c < cols && r < rows) out[(long long)c * rows + r] = f2bf(tile[tx][ty + i]);
  }
}

__global__ __launch_bounds__(256)
void convert_bf16_kernel(const float* __restrict__ in, bf16_t* __restrict__ out, long long n)
{
  const long long i = (long long)blockIdx.x * 256 + threadIdx.x;
  if (i < n) out[i] = f2bf(in[i]);
}

// POS_TABLE[l, d] for l<1024, d<512 (row 0 zeroed; even cols sin, odd cols cos)
__global__ __launch_bounds__(256)
void pos_table_kernel(float* __restrict__ pos)
{
  const int idx = blockIdx.x * 256 + threadIdx.x;
  if (idx >= LL * DD) return;
  const int l = idx >> 9, d = idx & 511;
  float val = 0.f;
  if (l > 0) {
    const float ang = (float)l * powf(10000.0f, -2.0f * (float)d / 512.0f);
    val = (d & 1) ? cosf(ang) : sinf(ang);
  }
  pos[idx] = val;
}

// ---------------------------------------------------------------------------
extern "C" void kernel_launch(void* const* d_in, const int* in_sizes, int n_in,
                              void* d_out, int out_size, void* d_ws, size_t ws_size,
                              hipStream_t stream)
{
  (void)in_sizes; (void)n_in; (void)out_size; (void)ws_size;

  const float* dec_inputs  = (const float*)d_in[0];
  const float* enc_outputs = (const float*)d_in[2];
  // params: jax pytree flatten order (dict keys sorted)
  const float* conv_b   = (const float*)d_in[3];
  const float* conv_w   = (const float*)d_in[4];    // [o=1024, c=1024]
  const float* dec_fc_b = (const float*)d_in[5];
  const float* dec_fc_w = (const float*)d_in[6];    // [512,256]
  const float* demb_b   = (const float*)d_in[7];
  const float* demb_w   = (const float*)d_in[8];    // [256,512]
  const float* k_b      = (const float*)d_in[9];
  const float* k_w      = (const float*)d_in[10];   // [512,512]
  struct Lw { const float *c_fc,*c_lnb,*c_lng,*c_wk,*c_wq,*c_wv,
                          *f_lnb,*f_lng,*f_w1,*f_w2,
                          *s_fc,*s_lnb,*s_lng,*s_wk,*s_wq,*s_wv; };
  Lw lw[2];
  for (int l = 0; l < 2; ++l) {
    const int o = 11 + l * 16;
    lw[l].c_fc  = (const float*)d_in[o + 0];  lw[l].c_lnb = (const float*)d_in[o + 1];
    lw[l].c_lng = (const float*)d_in[o + 2];  lw[l].c_wk  = (const float*)d_in[o + 3];
    lw[l].c_wq  = (const float*)d_in[o + 4];  lw[l].c_wv  = (const float*)d_in[o + 5];
    lw[l].f_lnb = (const float*)d_in[o + 6];  lw[l].f_lng = (const float*)d_in[o + 7];
    lw[l].f_w1  = (const float*)d_in[o + 8];  lw[l].f_w2  = (const float*)d_in[o + 9];
    lw[l].s_fc  = (const float*)d_in[o + 10]; lw[l].s_lnb = (const float*)d_in[o + 11];
    lw[l].s_lng = (const float*)d_in[o + 12]; lw[l].s_wk  = (const float*)d_in[o + 13];
    lw[l].s_wq  = (const float*)d_in[o + 14]; lw[l].s_wv  = (const float*)d_in[o + 15];
  }
  const float* q_b   = (const float*)d_in[43];
  const float* q_w   = (const float*)d_in[44];      // [512,512]
  const float* tgt_b = (const float*)d_in[45];
  const float* tgt_w = (const float*)d_in[46];      // [256,512]
  const float* v_b   = (const float*)d_in[47];
  const float* v_w   = (const float*)d_in[48];      // [512,512]

  // ---------------- workspace bump allocator ----------------
  char* cur = (char*)d_ws;
  auto alloc = [&](size_t bytes) -> char* {
    char* p = cur; cur += (bytes + 255) & ~(size_t)255; return p;
  };
  float*  pos_f      = (float*) alloc((size_t)LL * DD * 4);
  bf16_t* Wconv      = (bf16_t*)alloc((size_t)1024 * 1024 * 2);
  bf16_t* dec_fc_wt  = (bf16_t*)alloc((size_t)NFSZ * DD * 2);
  bf16_t* tgt_wt     = (bf16_t*)alloc((size_t)DD * NFSZ * 2);
  bf16_t* demb_wt    = (bf16_t*)alloc((size_t)DD * NFSZ * 2);
  bf16_t* q_wt       = (bf16_t*)alloc((size_t)DD * DD * 2);
  bf16_t* k_wt       = (bf16_t*)alloc((size_t)DD * DD * 2);
  bf16_t* v_wt       = (bf16_t*)alloc((size_t)DD * DD * 2);
  bf16_t *s_wq_t[2], *s_wk_t[2], *s_wv_t[2], *s_fc_t[2];
  bf16_t *c_wq_t[2], *c_wk_t[2], *c_wv_t[2], *c_fc_t[2];
  bf16_t *f_w1_t[2], *f_w2_t[2];
  for (int l = 0; l < 2; ++l) {
    s_wq_t[l] = (bf16_t*)alloc((size_t)64 * DD * 2);
    s_wk_t[l] = (bf16_t*)alloc((size_t)64 * DD * 2);
    s_wv_t[l] = (bf16_t*)alloc((size_t)64 * DD * 2);
    s_fc_t[l] = (bf16_t*)alloc((size_t)DD * 64 * 2);
    c_wq_t[l] = (bf16_t*)alloc((size_t)64 * DD * 2);
    c_wk_t[l] = (bf16_t*)alloc((size_t)64 * DD * 2);
    c_wv_t[l] = (bf16_t*)alloc((size_t)64 * DD * 2);
    c_fc_t[l] = (bf16_t*)alloc((size_t)DD * 64 * 2);
    f_w1_t[l] = (bf16_t*)alloc((size_t)DD * DD * 2);
    f_w2_t[l] = (bf16_t*)alloc((size_t)DD * DD * 2);
  }
  bf16_t* Ebf      = (bf16_t*)alloc((size_t)RR * DD * 2);        // enc_outputs bf16
  bf16_t* Ebt      = (bf16_t*)alloc((size_t)BB * DD * LL * 2);   // enc_outputs^T per batch
  bf16_t* decin_bf = (bf16_t*)alloc((size_t)RR * NFSZ * 2);
  bf16_t* d0_bf    = (bf16_t*)alloc((size_t)RR * DD * 2);        // dec_inputs0 (conv out)
  bf16_t* d0b_bf   = (bf16_t*)alloc((size_t)RR * NFSZ * 2);      // after dec_fc
  float*  dout0_f  = (float*) alloc((size_t)RR * DD * 4);        // dec_outputs0
  bf16_t* dout0_bf = (bf16_t*)alloc((size_t)RR * DD * 2);
  bf16_t* demb_bf  = (bf16_t*)alloc((size_t)RR * DD * 2);        // dec_in
  bf16_t* k_bf     = (bf16_t*)alloc((size_t)RR * DD * 2);
  float*  v_f      = (float*) alloc((size_t)RR * DD * 4);
  bf16_t* vt_bf    = (bf16_t*)alloc((size_t)BB * DD * LL * 2);
  float*  ca_f     = (float*) alloc((size_t)BB * LL * LL * 4);   // cross-attn scores (64MB)
  bf16_t* ca_bf    = (bf16_t*)alloc((size_t)BB * LL * LL * 2);
  float*  x_f      = (float*) alloc((size_t)RR * DD * 4);        // running stream
  bf16_t* x_bf     = (bf16_t*)alloc((size_t)RR * DD * 2);
  bf16_t* Qh_bf    = (bf16_t*)alloc((size_t)RR * 64 * 2);
  bf16_t* Kh_bf    = (bf16_t*)alloc((size_t)RR * 64 * 2);
  float*  Vh_f     = (float*) alloc((size_t)RR * 64 * 4);
  bf16_t* Vth_bf   = (bf16_t*)alloc((size_t)BB * 64 * LL * 2);
  float*  ctx_f    = (float*) alloc((size_t)RR * 64 * 4);
  bf16_t* ctx_bf   = (bf16_t*)alloc((size_t)RR * 64 * 2);
  // safe aliases (lifetimes disjoint):
  bf16_t* q_bf    = d0_bf;              // d0_bf dead after dec_fc GEMM
  bf16_t* attn_bf = (bf16_t*)ca_f;      // ca_f dead before layers (64MB == 64MB)
  float*  tmp_f   = v_f;                // v_f dead after transpose to vt_bf
  bf16_t* h_bf    = demb_bf;            // dec_in dead after q projection

  // ---------------- helpers ----------------
  auto convert = [&](const float* src, bf16_t* dst, long long n) {
    convert_bf16_kernel<<<dim3((unsigned)((n + 255) / 256)), dim3(256), 0, stream>>>(src, dst, n);
  };
  auto transpose = [&](const float* src, bf16_t* dst, int rows, int cols, int batch) {
    dim3 g((cols + 31) / 32, (rows + 31) / 32, batch);
    transpose_bf16_kernel<<<g, dim3(256), 0, stream>>>(
        src, (long long)rows * cols, dst, (long long)rows * cols, rows, cols);
  };
  auto gemm = [&](const bf16_t* A, int ldA, long long sAb, long long sAh,
                  const bf16_t* Bt, int ldB, long long sBb, long long sBh,
                  float* C, int ldC, long long sCb, long long sCh, bf16_t* Cb,
                  int M, int N, int K, int Z, int H, float alpha,
                  const float* biasN, const float* biasM,
                  const float* mulT, const float* addT,
                  const float* pos, int posL, int relu) {
    dim3 g((N + 127) / 128, (M + 127) / 128, Z);
    gemm_bf16_wmma<<<g, dim3(256), 0, stream>>>(
        A, ldA, sAb, sAh, Bt, ldB, sBb, sBh, C, ldC, sCb, sCh, Cb,
        M, N, K, H, alpha, biasN, biasM, mulT, addT, pos, posL, relu);
  };
  auto softmax = [&](float* x, bf16_t* xb, long long rows, int causal) {
    softmax1024<<<dim3((unsigned)rows), dim3(256), 0, stream>>>(x, xb, causal);
  };
  auto lnorm = [&](const float* x, const float* g, const float* b, float* y, bf16_t* yb) {
    layernorm512<<<dim3(RR), dim3(256), 0, stream>>>(x, g, b, y, yb);
  };

  const long long LD = (long long)LL * DD, L2v = (long long)LL * LL;
  const float invSqrtDk = 0.17677669529663687f;   // 1/sqrt(32)

  // ---------------- stage 0: tables & conversions ----------------
  pos_table_kernel<<<dim3((LL * DD + 255) / 256), dim3(256), 0, stream>>>(pos_f);
  convert(conv_w, Wconv, 1024LL * 1024);
  transpose(dec_fc_w, dec_fc_wt, DD, NFSZ, 1);
  transpose(tgt_w,  tgt_wt,  NFSZ, DD, 1);
  transpose(demb_w, demb_wt, NFSZ, DD, 1);
  transpose(q_w, q_wt, DD, DD, 1);
  transpose(k_w, k_wt, DD, DD, 1);
  transpose(v_w, v_wt, DD, DD, 1);
  for (int l = 0; l < 2; ++l) {
    transpose(lw[l].s_wq, s_wq_t[l], DD, 64, 1);
    transpose(lw[l].s_wk, s_wk_t[l], DD, 64, 1);
    transpose(lw[l].s_wv, s_wv_t[l], DD, 64, 1);
    transpose(lw[l].s_fc, s_fc_t[l], 64, DD, 1);
    transpose(lw[l].c_wq, c_wq_t[l], DD, 64, 1);
    transpose(lw[l].c_wk, c_wk_t[l], DD, 64, 1);
    transpose(lw[l].c_wv, c_wv_t[l], DD, 64, 1);
    transpose(lw[l].c_fc, c_fc_t[l], 64, DD, 1);
    transpose(lw[l].f_w1, f_w1_t[l], DD, DD, 1);
    transpose(lw[l].f_w2, f_w2_t[l], DD, DD, 1);
  }
  convert(enc_outputs, Ebf, (long long)RR * DD);
  transpose(enc_outputs, Ebt, LL, DD, BB);            // [b][512][1024]
  convert(dec_inputs, decin_bf, (long long)RR * NFSZ);

  // ---------------- stage 1: conv / embeddings ----------------
  // dec_inputs0[b,o,d] = conv_w[o,:] @ enc_outputs[b,:,d] + conv_b[o]
  gemm(Wconv, 1024, 0, 0, Ebt, LL, (long long)DD * LL, 0,
       nullptr, DD, LD, 0, d0_bf, LL, DD, LL, BB, 1, 1.f,
       nullptr, conv_b, nullptr, nullptr, nullptr, 1, 0);
  // @ dec_fc + b
  gemm(d0_bf, DD, 0, 0, dec_fc_wt, DD, 0, 0,
       nullptr, NFSZ, 0, 0, d0b_bf, RR, NFSZ, DD, 1, 1, 1.f,
       dec_fc_b, nullptr, nullptr, nullptr, nullptr, 1, 0);
  // dec_outputs0 = @ tgt + b + POS
  gemm(d0b_bf, NFSZ, 0, 0, tgt_wt, NFSZ, 0, 0,
       dout0_f, DD, 0, 0, dout0_bf, RR, DD, NFSZ, 1, 1, 1.f,
       tgt_b, nullptr, nullptr, nullptr, pos_f, LL, 0);
  // dec_in = dec_inputs @ demb + b
  gemm(decin_bf, NFSZ, 0, 0, demb_wt, NFSZ, 0, 0,
       nullptr, DD, 0, 0, demb_bf, RR, DD, NFSZ, 1, 1, 1.f,
       demb_b, nullptr, nullptr, nullptr, nullptr, 1, 0);
  // q,k,v projections
  gemm(demb_bf,  DD, 0, 0, q_wt, DD, 0, 0, nullptr, DD, 0, 0, q_bf,
       RR, DD, DD, 1, 1, 1.f, q_b, nullptr, nullptr, nullptr, nullptr, 1, 0);
  gemm(dout0_bf, DD, 0, 0, k_wt, DD, 0, 0, nullptr, DD, 0, 0, k_bf,
       RR, DD, DD, 1, 1, 1.f, k_b, nullptr, nullptr, nullptr, nullptr, 1, 0);
  gemm(dout0_bf, DD, 0, 0, v_wt, DD, 0, 0, v_f, DD, 0, 0, nullptr,
       RR, DD, DD, 1, 1, 1.f, v_b, nullptr, nullptr, nullptr, nullptr, 1, 0);
  transpose(v_f, vt_bf, LL, DD, BB);                  // [b][512][1024]

  // cross-attention quirk: softmax((q * sqrt(1024)) @ k^T) -> alpha = 32
  gemm(q_bf, DD, LD, 0, k_bf, DD, LD, 0,
       ca_f, LL, L2v, 0, nullptr, LL, LL, DD, BB, 1, 32.0f,
       nullptr, nullptr, nullptr, nullptr, nullptr, 1, 0);
  softmax(ca_f, ca_bf, (long long)BB * LL, 0);
  // dec_outputs = (ca @ v) * dec_outputs0 + POS
  gemm(ca_bf, LL, L2v, 0, vt_bf, LL, (long long)DD * LL, 0,
       x_f, DD, LD, 0, x_bf, LL, DD, LL, BB, 1, 1.f,
       nullptr, nullptr, dout0_f, nullptr, pos_f, LL, 0);

  // ---------------- stage 2: decoder layers ----------------
  float* base = (float*)d_out;
  const long long decElems  = (long long)RR * DD;                 // 8,388,608
  const long long attnElems = (long long)BB * 2 * LL * LL;        // 33,554,432

  for (int l = 0; l < 2; ++l) {
    // ---- self attention (causal) ----
    gemm(x_bf, DD, 0, 0, s_wq_t[l], DD, 0, 0, nullptr, 64, 0, 0, Qh_bf,
         RR, 64, DD, 1, 1, 1.f, nullptr, nullptr, nullptr, nullptr, nullptr, 1, 0);
    gemm(x_bf, DD, 0, 0, s_wk_t[l], DD, 0, 0, nullptr, 64, 0, 0, Kh_bf,
         RR, 64, DD, 1, 1, 1.f, nullptr, nullptr, nullptr, nullptr, nullptr, 1, 0);
    gemm(x_bf, DD, 0, 0, s_wv_t[l], DD, 0, 0, Vh_f, 64, 0, 0, nullptr,
         RR, 64, DD, 1, 1, 1.f, nullptr, nullptr, nullptr, nullptr, nullptr, 1, 0);
    transpose(Vh_f, Vth_bf, LL, 64, BB);              // [b][64][1024]
    float* selfC = base + decElems + (long long)l * attnElems;    // self_attns[l]
    gemm(Qh_bf, 64, (long long)LL * 64, 32, Kh_bf, 64, (long long)LL * 64, 32,
         selfC, LL, 2 * L2v, L2v, nullptr, LL, LL, 32, BB * 2, 2, invSqrtDk,
         nullptr, nullptr, nullptr, nullptr, nullptr, 1, 0);
    softmax(selfC, attn_bf, (long long)BB * 2 * LL, 1);
    gemm(attn_bf, LL, 2 * L2v, L2v, Vth_bf, LL, 64LL * LL, 32LL * LL,
         ctx_f, 64, (long long)LL * 64, 32, ctx_bf, LL, 32, LL, BB * 2, 2, 1.f,
         nullptr, nullptr, nullptr, nullptr, nullptr, 1, 0);
    gemm(ctx_bf, 64, 0, 0, s_fc_t[l], 64, 0, 0, tmp_f, DD, 0, 0, nullptr,
         RR, DD, 64, 1, 1, 1.f, nullptr, nullptr, x_f, nullptr, nullptr, 1, 0);
    lnorm(tmp_f, lw[l].s_lng, lw[l].s_lnb, x_f, x_bf);

    // ---- cross attention (no mask, K/V from enc_outputs) ----
    gemm(x_bf, DD, 0, 0, c_wq_t[l], DD, 0, 0, nullptr, 64, 0, 0, Qh_bf,
         RR, 64, DD, 1, 1, 1.f, nullptr, nullptr, nullptr, nullptr, nullptr, 1, 0);
    gemm(Ebf,  DD, 0, 0, c_wk_t[l], DD, 0, 0, nullptr, 64, 0, 0, Kh_bf,
         RR, 64, DD, 1, 1, 1.f, nullptr, nullptr, nullptr, nullptr, nullptr, 1, 0);
    gemm(Ebf,  DD, 0, 0, c_wv_t[l], DD, 0, 0, Vh_f, 64, 0, 0, nullptr,
         RR, 64, DD, 1, 1, 1.f, nullptr, nullptr, nullptr, nullptr, nullptr, 1, 0);
    transpose(Vh_f, Vth_bf, LL, 64, BB);
    float* encC = base + decElems + 2 * attnElems + (long long)l * attnElems; // enc_attns[l]
    gemm(Qh_bf, 64, (long long)LL * 64, 32, Kh_bf, 64, (long long)LL * 64, 32,
         encC, LL, 2 * L2v, L2v, nullptr, LL, LL, 32, BB * 2, 2, invSqrtDk,
         nullptr, nullptr, nullptr, nullptr, nullptr, 1, 0);
    softmax(encC, attn_bf, (long long)BB * 2 * LL, 0);
    gemm(attn_bf, LL, 2 * L2v, L2v, Vth_bf, LL, 64LL * LL, 32LL * LL,
         ctx_f, 64, (long long)LL * 64, 32, ctx_bf, LL, 32, LL, BB * 2, 2, 1.f,
         nullptr, nullptr, nullptr, nullptr, nullptr, 1, 0);
    gemm(ctx_bf, 64, 0, 0, c_fc_t[l], 64, 0, 0, tmp_f, DD, 0, 0, nullptr,
         RR, DD, 64, 1, 1, 1.f, nullptr, nullptr, x_f, nullptr, nullptr, 1, 0);
    lnorm(tmp_f, lw[l].c_lng, lw[l].c_lnb, x_f, x_bf);

    // ---- FFN: ln(relu(x@w1)@w2 + x) ----
    gemm(x_bf, DD, 0, 0, f_w1_t[l], DD, 0, 0, nullptr, DD, 0, 0, h_bf,
         RR, DD, DD, 1, 1, 1.f, nullptr, nullptr, nullptr, nullptr, nullptr, 1, 1);
    gemm(h_bf, DD, 0, 0, f_w2_t[l], DD, 0, 0, tmp_f, DD, 0, 0, nullptr,
         RR, DD, DD, 1, 1, 1.f, nullptr, nullptr, nullptr, x_f, nullptr, 1, 0);
    float* yOut = (l == 1) ? base : x_f;              // final layer -> d_out
    lnorm(tmp_f, lw[l].f_lng, lw[l].f_lnb, yOut, x_bf);
  }
}
